// NeuralLSV_47064251629761
// MI455X (gfx1250) — matrix operand order repaired and checked by reading, not compiled, not run on hardware
//
#include <hip/hip_runtime.h>
#include <math.h>

typedef __attribute__((ext_vector_type(16))) _Float16 v16h;
typedef __attribute__((ext_vector_type(8)))  _Float16 v8h;
typedef __attribute__((ext_vector_type(8)))  float    v8f;
typedef __attribute__((ext_vector_type(4)))  float    v4f;

// ---------------- problem constants ----------------
#define N_STEPS   32
#define N_MAT     4
#define N_STRIKES 21
#define BATCH     40000
#define PERIOD    8
#define DT        0.0625f
#define RFR       0.05f
#define SQRT_DT   0.25f            // sqrt(0.0625)
#define S0C       90.4837418036f   // 100*exp(-0.05*2)
#define BN_EPS    1e-5f

// ---------------- kernel tiling --------------------
#define B_TILE    160              // paths per block (10 WMMA col tiles)
#define NBLK      250              // 250*160 = 40000
#define NTHREADS  256
#define NWAVES    8

// ---------------- workspace layout (floats) --------
#define WS_PAYOFF_OFF   61440      // bn stats: [32][15][64][2] floats
#define WS_BAR_OFF      (61440 + 168)
#define WS_ZERO_WORDS   (61440 + 168 + 481)

// ---------------- shared memory layout (floats) ----
#define SM_ACTA   0            // f32 activations, 96 rows x 160 (padded)
#define SM_HEDGE  15360        // 84 x 160 hedging accumulators
#define SM_ACTB   28800        // f16 activations: 64 x 160 halves
#define SM_WBUF   33920        // f16 weights: 96 x 64 halves
#define SM_BIAS   36992        // 96 floats
#define SM_S      37088
#define SM_V      37248
#define SM_Z1     37408
#define SM_Z2     37568
#define SM_DS     37728
#define SM_BV     37888
#define SM_SV     38048
#define SM_RED    38208        // 256-entry reduction / BN-stat scratch
#define SMEM_FLOATS 38464      // ~150 KB (CDNA5: 320 KB LDS per workgroup)

struct NetParams {
    const float* W[5];
    const float* b[5];
    const float* g[5];
    const float* beta[5];
};
struct AllParams {
    NetParams sigmaS, bV, sigmaV, hedg;
    const float* rho;
    const float* V0;
    const float* Z;
    const float* strikes;
};

__device__ __forceinline__ float softplus_f(float x) {
    return (x > 20.f) ? x : log1pf(expf(x));
}

__device__ __forceinline__ void grid_barrier(int* bar, int id) {
    __syncthreads();
    if (threadIdx.x == 0) {
        __hip_atomic_fetch_add(&bar[id], 1, __ATOMIC_ACQ_REL, __HIP_MEMORY_SCOPE_AGENT);
        while (__hip_atomic_load(&bar[id], __ATOMIC_ACQUIRE, __HIP_MEMORY_SCOPE_AGENT) < NBLK) {
            __builtin_amdgcn_s_sleep(1);
        }
    }
    __syncthreads();
}

__device__ __forceinline__ v16h load_a_frag(const _Float16* __restrict__ Wbuf,
                                            int mt, int kt, int Kpad, int lane) {
    // 16-bit A 16x32 layout: lane<16 -> M=lane, K = {kb..kb+7, kb+16..kb+23};
    // lane>=16 -> same M with K base +8.
    const int r  = mt * 16 + (lane & 15);
    const int kb = kt * 32 + ((lane >> 4) << 3);
    const _Float16* wr = Wbuf + r * Kpad + kb;      // 16B aligned
    v8h lo = *(const v8h*)wr;
    v8h hi = *(const v8h*)(wr + 16);
    v16h a;
#pragma unroll
    for (int j = 0; j < 8; ++j) { a[j] = lo[j]; a[8 + j] = hi[j]; }
    return a;
}

__device__ __forceinline__ v16h load_b_frag(const _Float16* __restrict__ actB,
                                            int kt, int nt, int lane) {
    // B 32x16 layout: lane holds N=lane%16, K = {kb..kb+7, kb+16..kb+23}, kb = kt*32 + 8*(lane/16).
    const int kb = kt * 32 + ((lane >> 4) << 3);
    const _Float16* colp = actB + kb * B_TILE + nt * 16 + (lane & 15);
    v16h b;
#pragma unroll
    for (int j = 0; j < 8; ++j) { b[j] = colp[j * B_TILE]; b[8 + j] = colp[(16 + j) * B_TILE]; }
    return b;
}

// One layer: stage weights -> WMMA GEMM (bias in accumulator) -> optional full-batch BN.
template<int DIN, int DOUT, bool HAS_BN>
__device__ void net_layer(const float* __restrict__ Wg, const float* __restrict__ bg,
                          const float* __restrict__ gg, const float* __restrict__ betag,
                          int matIdx, int barId,
                          float* __restrict__ stats, int* __restrict__ bar,
                          float* __restrict__ actA, _Float16* __restrict__ actB,
                          _Float16* __restrict__ Wbuf, float* __restrict__ biasLds,
                          float* __restrict__ red,
                          int tid, int lane, int wave) {
    constexpr int Mt   = (DOUT + 15) / 16;
    constexpr int Kt   = (DIN + 31) / 32;
    constexpr int Kpad = Kt * 32;
    constexpr int KSH  = (Kpad == 64) ? 6 : 5;
    constexpr int ROWS = Mt * 16;

    // ---- stage weights (f32 global -> f16 LDS, zero padded) + bias ----
    const float* W = Wg + (size_t)matIdx * DOUT * DIN;
    for (int x = tid; x < ROWS * Kpad; x += NTHREADS) {
        const int r = x >> KSH, k = x & (Kpad - 1);
        Wbuf[x] = (_Float16)((r < DOUT && k < DIN) ? W[r * DIN + k] : 0.f);
    }
    for (int r = tid; r < ROWS; r += NTHREADS)
        biasLds[r] = (r < DOUT) ? bg[(size_t)matIdx * DOUT + r] : 0.f;
    __syncthreads();

    // ---- GEMM: row-group = wave>>1 (stride 4), col-half = wave&1 (5 tiles) ----
    const int nt0 = (wave & 1) * 5;
#pragma unroll
    for (int mg = 0; mg < (Mt + 3) / 4; ++mg) {
        const int mt = (wave >> 1) + mg * 4;
        if (mt < Mt) {
            v16h a0 = load_a_frag(Wbuf, mt, 0, Kpad, lane);
            v16h a1{};
            if constexpr (Kt == 2) a1 = load_a_frag(Wbuf, mt, 1, Kpad, lane);
            const int rbase = mt * 16 + ((lane >> 4) << 3);
            const float* bp = biasLds + rbase;                  // 32B aligned
            v4f blo = *(const v4f*)bp;
            v4f bhi = *(const v4f*)(bp + 4);
#pragma unroll
            for (int nc = 0; nc < 5; ++nc) {
                const int nt = nt0 + nc;
                v8f c;
                c[0] = blo[0]; c[1] = blo[1]; c[2] = blo[2]; c[3] = blo[3];
                c[4] = bhi[0]; c[5] = bhi[1]; c[6] = bhi[2]; c[7] = bhi[3];
                v16h b0 = load_b_frag(actB, 0, nt, lane);
                c = __builtin_amdgcn_wmma_f32_16x16x32_f16(false, a0, false, b0,
                                                           (short)0, c, false, false);
                if constexpr (Kt == 2) {
                    v16h b1 = load_b_frag(actB, 1, nt, lane);
                    c = __builtin_amdgcn_wmma_f32_16x16x32_f16(false, a1, false, b1,
                                                               (short)0, c, false, false);
                }
                // unconditional store into padded actA (pad rows never consumed)
                float* outp = actA + rbase * B_TILE + nt * 16 + (lane & 15);
#pragma unroll
                for (int rr = 0; rr < 8; ++rr) outp[rr * B_TILE] = c[rr];
            }
        }
    }
    __syncthreads();

    // ---- full-batch batch-norm + LeakyReLU ----
    if constexpr (HAS_BN) {
        float* st = stats + (size_t)barId * 128;   // 64 ch * {sum, sumsq}
        for (int x = tid; x < 128; x += NTHREADS) red[x] = 0.f;
        __syncthreads();
        {
            const int ch = tid >> 2;
            const int p0 = (tid & 3) * 40;
            if (ch < DOUT) {
                float s = 0.f, q = 0.f;
                const float* ap = actA + ch * B_TILE;
#pragma unroll 8
                for (int p = p0; p < p0 + 40; ++p) { float a = ap[p]; s += a; q += a * a; }
                atomicAdd(&red[ch * 2 + 0], s);   // ds_add_f32
                atomicAdd(&red[ch * 2 + 1], q);
            }
        }
        __syncthreads();
        if (tid < DOUT) {
            atomicAdd(&st[tid * 2 + 0], red[tid * 2 + 0]);
            atomicAdd(&st[tid * 2 + 1], red[tid * 2 + 1]);
        }
        grid_barrier(bar, barId);
        if (tid < DOUT) {
            const float s = st[tid * 2 + 0], q = st[tid * 2 + 1];
            const float mean = s * (1.f / (float)BATCH);
            const float var  = q * (1.f / (float)BATCH) - mean * mean;
            const float inv  = rsqrtf(var + BN_EPS);
            const float sc   = inv * gg[(size_t)matIdx * DOUT + tid];
            red[tid]      = sc;
            red[64 + tid] = betag[(size_t)matIdx * DOUT + tid] - mean * sc;
        }
        __syncthreads();
        constexpr int DPAD = ((DOUT + 31) / 32) * 32;
#pragma unroll
        for (int ch = wave; ch < DPAD; ch += NWAVES) {
            const bool cv = (ch < DOUT);
            const float sc = cv ? red[ch] : 0.f;
            const float sh = cv ? red[64 + ch] : 0.f;
            for (int pb = lane; pb < B_TILE / 2; pb += 32) {
                const int p = pb * 2;
                float y0 = cv ? actA[ch * B_TILE + p] * sc + sh : 0.f;
                float y1 = cv ? actA[ch * B_TILE + p + 1] * sc + sh : 0.f;
                y0 = (y0 > 0.f) ? y0 : 0.01f * y0;
                y1 = (y1 > 0.f) ? y1 : 0.01f * y1;
                union { _Float16 h[2]; unsigned u; } pk;
                pk.h[0] = (_Float16)y0; pk.h[1] = (_Float16)y1;
                *(unsigned*)(actB + ch * B_TILE + p) = pk.u;    // packed dword store
            }
        }
        __syncthreads();
    }
}

__global__ void init_ws_kernel(unsigned int* w, int n) {
    for (int i = blockIdx.x * blockDim.x + threadIdx.x; i < n; i += gridDim.x * blockDim.x)
        w[i] = 0u;
}

#define LAYER(NET, L, DIN, DOUT, BN, SLOT)                                          \
    net_layer<DIN, DOUT, BN>(NET.W[L], NET.b[L], NET.g[L], NET.beta[L], idx,        \
                             step * 15 + (SLOT), stats, bar, actA, actB, Wbuf,      \
                             biasLds, red, tid, lane, wave)

__global__ __launch_bounds__(NTHREADS)
void lsv_kernel(AllParams P, float* __restrict__ out, float* __restrict__ ws) {
    extern __shared__ float smem[];
    float*    actA    = smem + SM_ACTA;
    float*    hedge   = smem + SM_HEDGE;
    _Float16* actB    = (_Float16*)(smem + SM_ACTB);
    _Float16* Wbuf    = (_Float16*)(smem + SM_WBUF);
    float*    biasLds = smem + SM_BIAS;
    float*    Sarr    = smem + SM_S;
    float*    Varr    = smem + SM_V;
    float*    z1a     = smem + SM_Z1;
    float*    z2a     = smem + SM_Z2;
    float*    dSa     = smem + SM_DS;
    float*    bVa     = smem + SM_BV;
    float*    sVa     = smem + SM_SV;
    float*    red     = smem + SM_RED;

    float* stats  = ws;
    float* payoff = ws + WS_PAYOFF_OFF;
    int*   bar    = (int*)(ws + WS_BAR_OFF);

    const int tid  = threadIdx.x;
    const int lane = tid & 31;
    const int wave = tid >> 5;
    const int g0   = blockIdx.x * B_TILE;

    const float rho  = P.rho[0];
    const float V0v  = P.V0[0];
    const float rho2 = sqrtf(fmaxf(1.f - rho * rho, 0.f));

    for (int p = tid; p < B_TILE; p += NTHREADS) { Sarr[p] = S0C; Varr[p] = V0v; }
    for (int x = tid; x < N_STRIKES * N_MAT * B_TILE; x += NTHREADS) hedge[x] = 0.f;
    __syncthreads();

    for (int i = 1; i <= N_STEPS; ++i) {
        const int step = i - 1;
        const int idx  = step >> 3;          // (i-1)//PERIOD
        const float tval = (float)step * DT;

        for (int p = tid; p < B_TILE; p += NTHREADS) {
            z1a[p] = P.Z[(size_t)(0 * N_STEPS + step) * BATCH + g0 + p];
            z2a[p] = P.Z[(size_t)(1 * N_STEPS + step) * BATCH + g0 + p];
        }
        if (i < N_STEPS && tid == 0) {
            __builtin_prefetch(P.Z + (size_t)(step + 1) * BATCH + g0, 0, 1);
            __builtin_prefetch(P.Z + (size_t)(N_STEPS + step + 1) * BATCH + g0, 0, 1);
        }
        __syncthreads();

        // ---------------- sigma_S net: X = [t; S; V] ----------------
        for (int ch = wave; ch < 32; ch += NWAVES)
            for (int p = lane; p < B_TILE; p += 32) {
                float v = (ch == 0) ? tval : (ch == 1) ? Sarr[p] : (ch == 2) ? Varr[p] : 0.f;
                actB[ch * B_TILE + p] = (_Float16)v;
            }
        __syncthreads();
        LAYER(P.sigmaS, 0,  3, 50, true, 0);
        LAYER(P.sigmaS, 1, 50, 50, true, 1);
        LAYER(P.sigmaS, 2, 50, 50, true, 2);
        LAYER(P.sigmaS, 3, 50, 50, true, 3);
        LAYER(P.sigmaS, 4, 50,  1, false, 0);
        for (int p = tid; p < B_TILE; p += NTHREADS) dSa[p] = softplus_f(actA[p]);
        __syncthreads();

        // ---------------- b_V net: X = [V] ----------------
        for (int ch = wave; ch < 32; ch += NWAVES)
            for (int p = lane; p < B_TILE; p += 32)
                actB[ch * B_TILE + p] = (_Float16)((ch == 0) ? Varr[p] : 0.f);
        __syncthreads();
        LAYER(P.bV, 0,  1, 50, true, 4);
        LAYER(P.bV, 1, 50, 50, true, 5);
        LAYER(P.bV, 2, 50, 50, true, 6);
        LAYER(P.bV, 3, 50, 50, true, 7);
        LAYER(P.bV, 4, 50,  1, false, 0);
        for (int p = tid; p < B_TILE; p += NTHREADS) bVa[p] = actA[p];
        __syncthreads();

        // ---------------- sigma_V net: X = [V] ----------------
        for (int ch = wave; ch < 32; ch += NWAVES)
            for (int p = lane; p < B_TILE; p += 32)
                actB[ch * B_TILE + p] = (_Float16)((ch == 0) ? Varr[p] : 0.f);
        __syncthreads();
        LAYER(P.sigmaV, 0,  1, 50, true, 8);
        LAYER(P.sigmaV, 1, 50, 50, true, 9);
        LAYER(P.sigmaV, 2, 50, 50, true, 10);
        LAYER(P.sigmaV, 3, 50, 50, true, 11);
        LAYER(P.sigmaV, 4, 50,  1, false, 0);
        for (int p = tid; p < B_TILE; p += NTHREADS) sVa[p] = softplus_f(actA[p]);
        __syncthreads();

        // ---------------- hedging net: X = [t; S] ----------------
        for (int ch = wave; ch < 32; ch += NWAVES)
            for (int p = lane; p < B_TILE; p += 32) {
                float v = (ch == 0) ? tval : (ch == 1) ? Sarr[p] : 0.f;
                actB[ch * B_TILE + p] = (_Float16)v;
            }
        __syncthreads();
        LAYER(P.hedg, 0,  2, 30, true, 12);
        LAYER(P.hedg, 1, 30, 30, true, 13);
        LAYER(P.hedg, 2, 30, 30, true, 14);
        LAYER(P.hedg, 3, 30, N_STRIKES * N_MAT, false, 0);
        // actA rows 0..83 now hold hedge net output (row = strike*4 + mat)

        // ---------------- SDE step + hedging accumulation ----------------
        if (tid < B_TILE) {
            const int p = tid;
            const float Sp = Sarr[p], Vp = Varr[p], ds = dSa[p];
            const float bS = Sp * RFR / (1.f + fabsf(Sp * RFR) * SQRT_DT);
            const float sS = Sp * ds  / (1.f + fabsf(Sp * ds)  * SQRT_DT);
            const float dW = SQRT_DT * z2a[p];
            const float dB = rho * dW + rho2 * SQRT_DT * z1a[p];
            const float Snew = Sp + bS * DT + sS * dW;
            const float Vnew = fmaxf(Vp + bVa[p] * DT + sVa[p] * dB, 0.f);
            const float dfim1 = expf(-RFR * DT * (float)step);
            const float coef  = dfim1 * Sp * ds * dW;
#pragma unroll 4
            for (int row = 0; row < N_STRIKES * N_MAT; ++row)
                hedge[row * B_TILE + p] += coef * actA[row * B_TILE + p];
            Sarr[p] = Snew; Varr[p] = Vnew;
        }
        __syncthreads();

        // ---------------- maturity payoff stats ----------------
        if ((i & (PERIOD - 1)) == 0) {
            const float discount = expf(-RFR * DT * (float)i);
            for (int k = 0; k < N_STRIKES; ++k) {
                float sim = 0.f;
                if (tid < B_TILE) {
                    const float pay = fmaxf(Sarr[tid] - P.strikes[k], 0.f) * discount;
                    sim = pay - hedge[(k * N_MAT + idx) * B_TILE + tid];
                }
                red[tid] = sim; __syncthreads();
                for (int off = NTHREADS / 2; off > 0; off >>= 1) {
                    if (tid < off) red[tid] += red[tid + off];
                    __syncthreads();
                }
                if (tid == 0) atomicAdd(&payoff[(idx * N_STRIKES + k) * 2 + 0], red[0]);
                __syncthreads();
                red[tid] = sim * sim; __syncthreads();
                for (int off = NTHREADS / 2; off > 0; off >>= 1) {
                    if (tid < off) red[tid] += red[tid + off];
                    __syncthreads();
                }
                if (tid == 0) atomicAdd(&payoff[(idx * N_STRIKES + k) * 2 + 1], red[0]);
                __syncthreads();
            }
        }
    }

    // ---------------- finalize prices / variance ----------------
    grid_barrier(bar, N_STEPS * 15);
    if (blockIdx.x == 0 && tid < N_STRIKES * N_MAT) {
        const int k = tid / N_MAT, m = tid % N_MAT;
        const float s = payoff[(m * N_STRIKES + k) * 2 + 0];
        const float q = payoff[(m * N_STRIKES + k) * 2 + 1];
        const float mean = s * (1.f / (float)BATCH);
        const float var  = (q - s * mean) / ((float)BATCH - 1.f);
        out[k * N_MAT + m] = mean;                          // prices [21,4]
        out[N_STRIKES * N_MAT + k * N_MAT + m] = var;       // variance [21,4]
    }
}

extern "C" void kernel_launch(void* const* d_in, const int* in_sizes, int n_in,
                              void* d_out, int out_size, void* d_ws, size_t ws_size,
                              hipStream_t stream) {
    (void)in_sizes; (void)n_in; (void)out_size; (void)ws_size;
    // jax pytree flattening: top-level dict in setup order (params, rho, V0, Z, strikes),
    // params dict sorted by key: b_V, hedging, sigma_S, sigma_V; layer dicts sorted:
    // W, b, beta, g (hidden layers) / W, b (output layer).
    AllParams P{};
    int o = 0;
    auto grab = [&](NetParams& np, int nL) {
        for (int l = 0; l < nL; ++l) {
            np.W[l] = (const float*)d_in[o++];
            np.b[l] = (const float*)d_in[o++];
            if (l < nL - 1) {
                np.beta[l] = (const float*)d_in[o++];
                np.g[l]    = (const float*)d_in[o++];
            }
        }
    };
    grab(P.bV, 5);
    grab(P.hedg, 4);
    grab(P.sigmaS, 5);
    grab(P.sigmaV, 5);
    P.rho     = (const float*)d_in[o++];
    P.V0      = (const float*)d_in[o++];
    P.Z       = (const float*)d_in[o++];
    P.strikes = (const float*)d_in[o++];

    hipLaunchKernelGGL(init_ws_kernel, dim3(64), dim3(NTHREADS), 0, stream,
                       (unsigned int*)d_ws, (int)WS_ZERO_WORDS);

    const size_t shmem = (size_t)SMEM_FLOATS * sizeof(float);   // ~150 KB (CDNA5: 320 KB/WG)
    hipLaunchKernelGGL(lsv_kernel, dim3(NBLK), dim3(NTHREADS), shmem, stream,
                       P, (float*)d_out, (float*)d_ws);
}
